// MultiAgentGNN_43327630082418
// MI455X (gfx1250) — matrix operand order repaired and checked by reading, not compiled, not run on hardware
//
#include <hip/hip_runtime.h>

typedef __attribute__((ext_vector_type(16))) _Float16 v16h;
typedef __attribute__((ext_vector_type(8)))  _Float16 v8h;
typedef __attribute__((ext_vector_type(8)))  float    v8f;

#define WAVES 4
#define SX 40    // x staging row stride (halves), 80B -> conflict-free b128 A loads
#define SA 136   // activation row stride (halves), 272B
#define S4 132   // fp32 buffer row stride (floats), 528B

// packed-weight offsets in d_ws (units: halves). frag = 512 halves (32 lanes x 16)
#define OFF_WIN 0L        // KT=1, NT=8   (K=30 padded to 32, N=128)
#define OFF_G1  4096L     // KT=4, NT=8
#define OFF_WV  20480L
#define OFF_WSK 36864L
#define OFF_G2  53248L
#define OFF_H1  69632L
#define OFF_H2  86016L    // KT=4, NT=4  (N=64)
#define OFF_H3  94208L    // KT=2, NT=32 (K=64, N=512)
// total = 126976 halves = 253,952 bytes of ws

__device__ __forceinline__ float gelu_erf(float x) {
  return 0.5f * x * (1.0f + erff(x * 0.70710678118654752440f));
}

__device__ __forceinline__ v8f wmma_f16(v16h a, v16h b, v8f c) {
  return __builtin_amdgcn_wmma_f32_16x16x32_f16(false, a, false, b, (short)0, c,
                                                false, false);
}

// A fragment (16x32 f16): lane L<16 holds row L, K in {k0..k0+7, k0+16..k0+23};
// lane L>=16 holds row L-16, K in {k0+8..k0+15, k0+24..k0+31}.
__device__ __forceinline__ v16h loadA(const _Float16* base, int strideH, int kt, int lane) {
  int m = lane & 15, hi = lane >> 4;
  const _Float16* p = base + m * strideH + kt * 32 + hi * 8;
  v8h a0 = *(const v8h*)(p);
  v8h a1 = *(const v8h*)(p + 16);
  v16h a;
#pragma unroll
  for (int i = 0; i < 8; ++i) { a[i] = a0[i]; a[i + 8] = a1[i]; }
  return a;
}

// GEMM: [16 x 32*KT] (A in LDS, f16) x [32*KT x 16*NT] (B prepacked in ws)
// writes fp32 (acc + bias[col]) into obuf.
template <int KT, int NT>
__device__ __forceinline__ void gemm_store(const _Float16* A, int strideA,
                                           const _Float16* W, float* obuf,
                                           const float* __restrict__ bias, int lane) {
  int m = lane & 15, hi = lane >> 4;
  for (int nt = 0; nt < NT; ++nt) {
    v8f acc = {};
#pragma unroll
    for (int kt = 0; kt < KT; ++kt) {
      v16h a = loadA(A, strideA, kt, lane);
      v16h b = ((const v16h*)W)[(nt * KT + kt) * 32 + lane];
      acc = wmma_f16(a, b, acc);
    }
    int col = nt * 16 + m;
    float bb = bias[col];
#pragma unroll
    for (int r = 0; r < 8; ++r) obuf[(r + 8 * hi) * S4 + col] = acc[r] + bb;
  }
}

__global__ void pack_w(const float* __restrict__ src, _Float16* __restrict__ dst,
                       int Kreal, int KT, int NT, int Nfull) {
  int f = blockIdx.x * blockDim.x + threadIdx.x;
  int total = KT * NT * 512;
  if (f >= total) return;
  int frag = f >> 9, rem = f & 511;
  int lane = rem >> 4, h = rem & 15;
  int nt = frag / KT, kt = frag - nt * KT;
  int k = kt * 32 + ((h >> 3) << 4) + ((lane >> 4) << 3) + (h & 7);  // B layout: lanes 0-15 K=0..7,16..23; 16-31 K=8..15,24..31
  int n = nt * 16 + (lane & 15);
  float v = (k < Kreal) ? src[k * Nfull + n] : 0.0f;
  dst[f] = (_Float16)v;
}

__global__ __launch_bounds__(WAVES * 32) void gnn_fused(
    const float* __restrict__ x, const _Float16* __restrict__ ws,
    const float* __restrict__ b_in, const float* __restrict__ ln0g, const float* __restrict__ ln0b,
    const float* __restrict__ g1b,  const float* __restrict__ ln1g, const float* __restrict__ ln1b,
    const float* __restrict__ bv,   const float* __restrict__ bsk,
    const float* __restrict__ ln2g, const float* __restrict__ ln2b,
    const float* __restrict__ g2b,  const float* __restrict__ ln3g, const float* __restrict__ ln3b,
    const float* __restrict__ hb1,  const float* __restrict__ hb2,  const float* __restrict__ hb3,
    float* __restrict__ out) {
  __shared__ _Float16 s_x[WAVES * 16 * SX];
  __shared__ _Float16 s_act[WAVES * 16 * SA];
  __shared__ float    s_f32[WAVES * 16 * S4];

  const int lane = threadIdx.x & 31;
  const int wave = threadIdx.x >> 5;
  _Float16* xb  = s_x   + wave * 16 * SX;
  _Float16* act = s_act + wave * 16 * SA;
  float*    fb  = s_f32 + wave * 16 * S4;
  const int rowBase = (blockIdx.x * WAVES + wave) * 16;
  const int m = lane & 15, hi = lane >> 4;

  // ---- stage x tile [16 x 30] -> f16 LDS [16 x 32] (cols 30,31 zero) ----
  xb[m * SX + 30 + hi] = (_Float16)0.0f;
#pragma unroll
  for (int i = 0; i < 15; ++i) {
    int idx = lane + 32 * i;           // 0..479
    int r = idx / 30, c = idx - r * 30;
    xb[r * SX + c] = (_Float16)x[(size_t)(rowBase + r) * 30 + c];
  }

  // ---- L1: h0 = LN0(GELU(x@w_in + b_in)) ----
  gemm_store<1, 8>(xb, SX, ws + OFF_WIN, fb, b_in, lane);
  {
    float* p = fb + m * S4 + hi * 64;
    float s = 0.f, s2 = 0.f;
#pragma unroll 4
    for (int c = 0; c < 64; ++c) { float v = gelu_erf(p[c]); p[c] = v; s += v; s2 += v * v; }
    s += __shfl_xor(s, 16, 32); s2 += __shfl_xor(s2, 16, 32);
    float mean = s * 0.0078125f, rs = rsqrtf(s2 * 0.0078125f - mean * mean + 1e-5f);
#pragma unroll 4
    for (int c = 0; c < 64; ++c) { int cc = hi * 64 + c;
      act[m * SA + cc] = (_Float16)((p[c] - mean) * rs * ln0g[cc] + ln0b[cc]); }
  }

  // ---- L2: h1 = LN1(relu(h0@gat1_w + gat1_b)) ----
  gemm_store<4, 8>(act, SA, ws + OFF_G1, fb, g1b, lane);
  {
    float* p = fb + m * S4 + hi * 64;
    float s = 0.f, s2 = 0.f;
#pragma unroll 4
    for (int c = 0; c < 64; ++c) { float v = fmaxf(p[c], 0.f); p[c] = v; s += v; s2 += v * v; }
    s += __shfl_xor(s, 16, 32); s2 += __shfl_xor(s2, 16, 32);
    float mean = s * 0.0078125f, rs = rsqrtf(s2 * 0.0078125f - mean * mean + 1e-5f);
#pragma unroll 4
    for (int c = 0; c < 64; ++c) { int cc = hi * 64 + c;
      act[m * SA + cc] = (_Float16)((p[c] - mean) * rs * ln1g[cc] + ln1b[cc]); }
  }

  // ---- L3: h2 = LN2((h1@wv + bv) + (h1@wskip + bskip)) (dual GEMM, shared A) ----
  for (int nt = 0; nt < 8; ++nt) {
    v8f av = {}, as = {};
#pragma unroll
    for (int kt = 0; kt < 4; ++kt) {
      v16h a  = loadA(act, SA, kt, lane);
      v16h b0 = ((const v16h*)(ws + OFF_WV ))[(nt * 4 + kt) * 32 + lane];
      v16h b1 = ((const v16h*)(ws + OFF_WSK))[(nt * 4 + kt) * 32 + lane];
      av = wmma_f16(a, b0, av);
      as = wmma_f16(a, b1, as);
    }
    int col = nt * 16 + m;
    float c0 = bv[col], c1 = bsk[col];
#pragma unroll
    for (int r = 0; r < 8; ++r) fb[(r + 8 * hi) * S4 + col] = (av[r] + c0) + (as[r] + c1);
  }
  {
    float* p = fb + m * S4 + hi * 64;
    float s = 0.f, s2 = 0.f;
#pragma unroll 4
    for (int c = 0; c < 64; ++c) { float v = p[c]; s += v; s2 += v * v; }
    s += __shfl_xor(s, 16, 32); s2 += __shfl_xor(s2, 16, 32);
    float mean = s * 0.0078125f, rs = rsqrtf(s2 * 0.0078125f - mean * mean + 1e-5f);
#pragma unroll 4
    for (int c = 0; c < 64; ++c) { int cc = hi * 64 + c;
      act[m * SA + cc] = (_Float16)((p[c] - mean) * rs * ln2g[cc] + ln2b[cc]); }
  }

  // ---- L4: h3 = LN3(h2@gat2_w + gat2_b) ----
  gemm_store<4, 8>(act, SA, ws + OFF_G2, fb, g2b, lane);
  {
    float* p = fb + m * S4 + hi * 64;
    float s = 0.f, s2 = 0.f;
#pragma unroll 4
    for (int c = 0; c < 64; ++c) { float v = p[c]; s += v; s2 += v * v; }
    s += __shfl_xor(s, 16, 32); s2 += __shfl_xor(s2, 16, 32);
    float mean = s * 0.0078125f, rs = rsqrtf(s2 * 0.0078125f - mean * mean + 1e-5f);
#pragma unroll 4
    for (int c = 0; c < 64; ++c) { int cc = hi * 64 + c;
      act[m * SA + cc] = (_Float16)((p[c] - mean) * rs * ln3g[cc] + ln3b[cc]); }
  }

  // ---- L5: g1 = GELU(h3@head_w1 + head_b1) ----
  gemm_store<4, 8>(act, SA, ws + OFF_H1, fb, hb1, lane);
#pragma unroll 4
  for (int c = 0; c < 64; ++c) {
    int cc = hi * 64 + c;
    act[m * SA + cc] = (_Float16)gelu_erf(fb[m * S4 + cc]);
  }

  // ---- L6: g2 = GELU(g1@head_w2 + head_b2), 64 cols ----
  gemm_store<4, 4>(act, SA, ws + OFF_H2, fb, hb2, lane);
#pragma unroll 4
  for (int c = 0; c < 32; ++c) {
    int cc = hi * 32 + c;
    act[m * SA + cc] = (_Float16)gelu_erf(fb[m * S4 + cc]);
  }

  // ---- L7: out = g2@head_w3 + head_b3, [16 x 512] straight to HBM ----
  for (int nt = 0; nt < 32; ++nt) {
    v8f acc = {};
#pragma unroll
    for (int kt = 0; kt < 2; ++kt) {
      v16h a = loadA(act, SA, kt, lane);
      v16h b = ((const v16h*)(ws + OFF_H3))[(nt * 2 + kt) * 32 + lane];
      acc = wmma_f16(a, b, acc);
    }
    int col = nt * 16 + m;
    float bb = hb3[col];
#pragma unroll
    for (int r = 0; r < 8; ++r)
      out[(size_t)(rowBase + r + 8 * hi) * 512 + col] = acc[r] + bb;
  }
}

extern "C" void kernel_launch(void* const* d_in, const int* in_sizes, int n_in,
                              void* d_out, int out_size, void* d_ws, size_t ws_size,
                              hipStream_t stream) {
  const float* x     = (const float*)d_in[0];
  const float* w_in  = (const float*)d_in[1];
  const float* b_in  = (const float*)d_in[2];
  const float* ln0g  = (const float*)d_in[3];
  const float* ln0b  = (const float*)d_in[4];
  const float* gat1w = (const float*)d_in[5];
  const float* gat1b = (const float*)d_in[8];
  const float* ln1g  = (const float*)d_in[9];
  const float* ln1b  = (const float*)d_in[10];
  const float* wv    = (const float*)d_in[15];
  const float* bvv   = (const float*)d_in[16];
  const float* wskip = (const float*)d_in[17];
  const float* bskip = (const float*)d_in[18];
  const float* ln2g  = (const float*)d_in[19];
  const float* ln2b  = (const float*)d_in[20];
  const float* gat2w = (const float*)d_in[21];
  const float* gat2b = (const float*)d_in[24];
  const float* ln3g  = (const float*)d_in[25];
  const float* ln3b  = (const float*)d_in[26];
  const float* hw1   = (const float*)d_in[27];
  const float* hb1   = (const float*)d_in[28];
  const float* hw2   = (const float*)d_in[29];
  const float* hb2   = (const float*)d_in[30];
  const float* hw3   = (const float*)d_in[31];
  const float* hb3   = (const float*)d_in[32];
  _Float16* ws = (_Float16*)d_ws;

  // Pre-swizzle weights (f32 row-major -> f16 WMMA B-fragment layout in ws).
  auto pk = [&](const float* src, long off, int Kreal, int KT, int NT, int Nf) {
    int total = KT * NT * 512;
    pack_w<<<(total + 255) / 256, 256, 0, stream>>>(src, ws + off, Kreal, KT, NT, Nf);
  };
  pk(w_in,  OFF_WIN, 30,  1, 8,  128);
  pk(gat1w, OFF_G1,  128, 4, 8,  128);
  pk(wv,    OFF_WV,  128, 4, 8,  128);
  pk(wskip, OFF_WSK, 128, 4, 8,  128);
  pk(gat2w, OFF_G2,  128, 4, 8,  128);
  pk(hw1,   OFF_H1,  128, 4, 8,  128);
  pk(hw2,   OFF_H2,  128, 4, 4,  64);
  pk(hw3,   OFF_H3,  64,  2, 32, 512);

  // 65536 rows / (16 rows * 4 waves) = 1024 blocks of 128 threads.
  gnn_fused<<<1024, WAVES * 32, 0, stream>>>(
      x, ws, b_in, ln0g, ln0b, gat1b, ln1g, ln1b, bvv, bskip, ln2g, ln2b,
      gat2b, ln3g, ln3b, hb1, hb2, hb3, (float*)d_out);
}
